// VisionAttention_72997264163464
// MI455X (gfx1250) — matrix operand order repaired and checked by reading, not compiled, not run on hardware
//
#include <hip/hip_runtime.h>
#include <hip/hip_bf16.h>

// ---------------------------------------------------------------------------
// VisionAttention for MI455X (gfx1250): bf16 WMMA everywhere, fp32 accumulate.
// Compute-bound (~52 GFLOP, working set fits in 192MB L2) -> keep v_wmma fed:
// 2-stage register ping-pong pipelines on all GEMM K-loops, P stored as bf16
// in LDS so the PV loop does no per-element converts.
// ---------------------------------------------------------------------------

typedef __attribute__((ext_vector_type(16))) __bf16 v16bf;
typedef __attribute__((ext_vector_type(8)))  __bf16 v8bf;
typedef __attribute__((ext_vector_type(8)))  float  v8f;

#define S_TOT 3072
#define EMB   1280
#define NH    16
#define HD    80
#define HDP   96     // head dim padded to multiple of 32 for WMMA K-loops
#define SEG   768
#define NSEG  4

// A-fragment in contiguous k order, not the ISA table order (16-bit A 16x32, wave32):
//   lane L (row M = L&15, h = L>>4): elems 0..15 -> K = h*16+j (matches B loads)
__device__ __forceinline__ v16bf load_afrag(const __bf16* rowk, int hh) {
  v8bf lo = *(const v8bf*)(rowk + hh * 16);
  v8bf hi = *(const v8bf*)(rowk + hh * 16 + 8);
  return __builtin_shufflevector(lo, hi, 0, 1, 2, 3, 4, 5, 6, 7,
                                 8, 9, 10, 11, 12, 13, 14, 15);
}

// ---------------------------------------------------------------------------
// Kernel 0: fp32 -> bf16 conversion (grid-stride)
// ---------------------------------------------------------------------------
__global__ __launch_bounds__(256) void cvt_bf16_kernel(const float* __restrict__ src,
                                                       __bf16* __restrict__ dst, int n) {
  for (int i = blockIdx.x * blockDim.x + threadIdx.x; i < n; i += gridDim.x * blockDim.x)
    dst[i] = (__bf16)src[i];
}

// ---------------------------------------------------------------------------
// Generic GEMM: C[M,N] = A[M,K] * B[N,K]^T + bias[N]   (K % 64 == 0)
// Block = 256 threads (8 waves). Block tile 64(M) x 128(N).
// Each wave: one 16-wide N strip, 4 stacked 16x16 C tiles.
// Explicit 2-stage ping-pong: fragments for k+32 in flight while WMMAs for k issue.
// ---------------------------------------------------------------------------
template <bool OUT_F32>
__global__ __launch_bounds__(256) void gemm_bf16_kernel(
    const __bf16* __restrict__ A, const __bf16* __restrict__ B,
    const float* __restrict__ bias, void* __restrict__ C,
    int M, int N, int K) {
  const int tid  = threadIdx.x;
  const int wave = tid >> 5;
  const int lane = tid & 31;
  const int lr   = lane & 15;
  const int hh   = lane >> 4;

  const int m0 = blockIdx.y * 64;
  const int n  = blockIdx.x * 128 + wave * 16 + lr;

  const __bf16* brow = B + (size_t)n * K;
  const __bf16* arow[4];
#pragma unroll
  for (int mt = 0; mt < 4; ++mt)
    arow[mt] = A + (size_t)(m0 + mt * 16 + lr) * K;

  v8f acc[4] = {{}, {}, {}, {}};
  v16bf bA, bB, aA[4], aB[4];

  bA = *(const v16bf*)(brow + hh * 16);
#pragma unroll
  for (int mt = 0; mt < 4; ++mt) aA[mt] = load_afrag(arow[mt], hh);

  for (int k0 = 0; k0 < K; k0 += 64) {
    __builtin_prefetch(brow + k0 + 256, 0, 1);  // global_prefetch_b8
    // stage: load k0+32 while computing k0
    bB = *(const v16bf*)(brow + k0 + 32 + hh * 16);
#pragma unroll
    for (int mt = 0; mt < 4; ++mt) aB[mt] = load_afrag(arow[mt] + k0 + 32, hh);
#pragma unroll
    for (int mt = 0; mt < 4; ++mt)
      acc[mt] = __builtin_amdgcn_wmma_f32_16x16x32_bf16(
          false, aA[mt], false, bA, (short)0, acc[mt], false, false);
    // stage: load k0+64 while computing k0+32
    if (k0 + 64 < K) {
      bA = *(const v16bf*)(brow + k0 + 64 + hh * 16);
#pragma unroll
      for (int mt = 0; mt < 4; ++mt) aA[mt] = load_afrag(arow[mt] + k0 + 64, hh);
    }
#pragma unroll
    for (int mt = 0; mt < 4; ++mt)
      acc[mt] = __builtin_amdgcn_wmma_f32_16x16x32_bf16(
          false, aB[mt], false, bB, (short)0, acc[mt], false, false);
  }

  const float bv = bias[n];
#pragma unroll
  for (int mt = 0; mt < 4; ++mt) {
#pragma unroll
    for (int i = 0; i < 8; ++i) {
      const int m = m0 + mt * 16 + hh * 8 + i;
      const float v = acc[mt][i] + bv;
      if (OUT_F32)
        ((float*)C)[(size_t)m * N + n] = v;
      else
        ((__bf16*)C)[(size_t)m * N + n] = (__bf16)v;
    }
  }
}

// ---------------------------------------------------------------------------
// RoPE + layout transform.
// qkv_bf: [S, 3*EMB] bf16.  cos/sin: [S, HD] fp32.
// Outputs: qh,kh: [NH][S][HDP] bf16 (cols HD..HDP-1 zero);
//          vT:    [NH][HDP][S] bf16 (rows HD..HDP-1 zero)  -> B operand of P*V.
// ---------------------------------------------------------------------------
__global__ __launch_bounds__(256) void rope_kernel(
    const __bf16* __restrict__ qkv, const float* __restrict__ cosb,
    const float* __restrict__ sinb, __bf16* __restrict__ qh,
    __bf16* __restrict__ kh, __bf16* __restrict__ vT) {
  int idx = blockIdx.x * blockDim.x + threadIdx.x;  // S * NH * HDP work items
  const int d = idx % HDP;
  int t = idx / HDP;
  const int h = t % NH;
  const int s = t / NH;
  if (s >= S_TOT) return;

  const size_t qo = (size_t)h * S_TOT * HDP + (size_t)s * HDP + d;
  const size_t vo = (size_t)h * HDP * S_TOT + (size_t)d * S_TOT + s;
  if (d < HD) {
    const size_t rbase = (size_t)s * (3 * EMB);
    const int e = h * HD + d;
    const float cv = cosb[s * HD + d];
    const float sv = sinb[s * HD + d];
    const int dro = (d < HD / 2) ? d + HD / 2 : d - HD / 2;
    const float sgn = (d < HD / 2) ? -1.f : 1.f;
    const float qv = (float)qkv[rbase + e];
    const float qr = (float)qkv[rbase + h * HD + dro];
    const float kv = (float)qkv[rbase + EMB + e];
    const float kr = (float)qkv[rbase + EMB + h * HD + dro];
    const float vv = (float)qkv[rbase + 2 * EMB + e];
    qh[qo] = (__bf16)(qv * cv + sgn * qr * sv);
    kh[qo] = (__bf16)(kv * cv + sgn * kr * sv);
    vT[vo] = (__bf16)vv;
  } else {
    qh[qo] = (__bf16)0.f;
    kh[qo] = (__bf16)0.f;
    vT[vo] = (__bf16)0.f;
  }
}

// ---------------------------------------------------------------------------
// Attention: one workgroup per (head, segment, 16-row q tile).
// Phase 1 : S = scale * Q Kt      (WMMA, K=HDP in 3 steps) -> LDS [16][SEG] f32
// Phase 2 : row max/sum (shuffle reductions), exp in LDS, 1/sum per row
// Phase 2b: compact P to bf16 in place (normalization folded into the pack)
// Phase 3 : O = P V               (WMMA, K=SEG, pipelined), write bf16 [s][e]
// ---------------------------------------------------------------------------
__global__ __launch_bounds__(256) void attn_kernel(
    const __bf16* __restrict__ qh, const __bf16* __restrict__ kh,
    const __bf16* __restrict__ vT, __bf16* __restrict__ attn) {
  __shared__ float sc[16 * SEG];  // 48 KB (also holds bf16 P in its first half)
  __shared__ float rowinv[16];

  const int tid  = threadIdx.x;
  const int wave = tid >> 5;
  const int lane = tid & 31;
  const int lr   = lane & 15;
  const int hh   = lane >> 4;

  int b = blockIdx.x;
  const int qt = b % (SEG / 16); b /= (SEG / 16);
  const int g  = b % NSEG;       b /= NSEG;
  const int h  = b;
  const int s0 = g * SEG;
  const int q0 = s0 + qt * 16;
  const float scale = 0.11180339887498949f;  // 80^-0.5

  // ---- Phase 1: scores ----
  v16bf aq[3];
  {
    const __bf16* abase = qh + (size_t)h * S_TOT * HDP + (size_t)(q0 + lr) * HDP;
#pragma unroll
    for (int kb = 0; kb < 3; ++kb) aq[kb] = load_afrag(abase + kb * 32, hh);
  }
  for (int kt = wave * 6; kt < wave * 6 + 6; ++kt) {
    v8f acc = {};
    const __bf16* bbase =
        kh + (size_t)h * S_TOT * HDP + (size_t)(s0 + kt * 16 + lr) * HDP;
#pragma unroll
    for (int kb = 0; kb < 3; ++kb) {
      v16bf bfrag = *(const v16bf*)(bbase + kb * 32 + hh * 16);
      acc = __builtin_amdgcn_wmma_f32_16x16x32_bf16(
          false, aq[kb], false, bfrag, (short)0, acc, false, false);
    }
#pragma unroll
    for (int i = 0; i < 8; ++i)
      sc[(hh * 8 + i) * SEG + kt * 16 + lr] = acc[i] * scale;
  }
  __syncthreads();

  // ---- Phase 2: softmax statistics + exp ----
  {
    const int row = tid >> 4;   // 0..15
    const int sub = tid & 15;
    float mx = -1e30f;
    for (int c = sub; c < SEG; c += 16) mx = fmaxf(mx, sc[row * SEG + c]);
#pragma unroll
    for (int o = 8; o > 0; o >>= 1) mx = fmaxf(mx, __shfl_xor(mx, o, 32));
    float sum = 0.f;
    for (int c = sub; c < SEG; c += 16) {
      float e = __expf(sc[row * SEG + c] - mx);
      sc[row * SEG + c] = e;
      sum += e;
    }
#pragma unroll
    for (int o = 8; o > 0; o >>= 1) sum += __shfl_xor(sum, o, 32);
    if (sub == 0) rowinv[row] = 1.f / sum;
  }
  __syncthreads();

  // ---- Phase 2b: compact fp32 P -> bf16 in place (read-all, barrier, write) ----
  {
    unsigned tmp[24];                       // 16*SEG/2 pairs / 256 threads = 24
#pragma unroll
    for (int it = 0; it < 24; ++it) {
      const int j = tid + it * 256;         // pair index
      const float inv = rowinv[j / (SEG / 2)];
      union { __bf16 h2[2]; unsigned u; } u;
      u.h2[0] = (__bf16)(sc[2 * j] * inv);
      u.h2[1] = (__bf16)(sc[2 * j + 1] * inv);
      tmp[it] = u.u;
    }
    __syncthreads();
#pragma unroll
    for (int it = 0; it < 24; ++it)
      ((unsigned*)sc)[tid + it * 256] = tmp[it];
  }
  __syncthreads();

  // ---- Phase 3: O = P V  (P bf16 from LDS, V pipelined from global) ----
  if (wave < HD / 16) {  // 5 d-tiles
    const __bf16* pb = (const __bf16*)sc;   // [16][SEG] bf16
    const int d = wave * 16 + lr;
    const __bf16* vb = vT + (size_t)h * HDP * S_TOT + (size_t)d * S_TOT + s0;

    v8f acc = {};
    v16bf vA = *(const v16bf*)(vb + hh * 16);
    for (int kk = 0; kk < SEG; kk += 64) {
      v16bf vB = *(const v16bf*)(vb + kk + 32 + hh * 16);
      v16bf af = load_afrag(pb + lr * SEG + kk, hh);
      acc = __builtin_amdgcn_wmma_f32_16x16x32_bf16(
          false, af, false, vA, (short)0, acc, false, false);
      if (kk + 64 < SEG) vA = *(const v16bf*)(vb + kk + 64 + hh * 16);
      v16bf af2 = load_afrag(pb + lr * SEG + kk + 32, hh);
      acc = __builtin_amdgcn_wmma_f32_16x16x32_bf16(
          false, af2, false, vB, (short)0, acc, false, false);
    }
#pragma unroll
    for (int i = 0; i < 8; ++i) {
      const int r = q0 + hh * 8 + i;
      attn[(size_t)r * EMB + h * HD + wave * 16 + lr] = (__bf16)acc[i];
    }
  }
}

// ---------------------------------------------------------------------------
// Launch
// ---------------------------------------------------------------------------
extern "C" void kernel_launch(void* const* d_in, const int* in_sizes, int n_in,
                              void* d_out, int out_size, void* d_ws, size_t ws_size,
                              hipStream_t stream) {
  const float* x      = (const float*)d_in[0];
  const float* cosb   = (const float*)d_in[1];
  const float* sinb   = (const float*)d_in[2];
  const float* w_qkv  = (const float*)d_in[3];
  const float* b_qkv  = (const float*)d_in[4];
  const float* w_proj = (const float*)d_in[5];
  const float* b_proj = (const float*)d_in[6];
  (void)in_sizes; (void)n_in; (void)ws_size; (void)out_size;

  // workspace carve-up (bf16 elements)
  char* p = (char*)d_ws;
  __bf16* xbf   = (__bf16*)p; p += (size_t)S_TOT * EMB * 2;
  __bf16* wqkvb = (__bf16*)p; p += (size_t)3 * EMB * EMB * 2;
  __bf16* wprjb = (__bf16*)p; p += (size_t)EMB * EMB * 2;
  __bf16* qkvb  = (__bf16*)p; p += (size_t)S_TOT * 3 * EMB * 2;
  __bf16* qhb   = (__bf16*)p; p += (size_t)NH * S_TOT * HDP * 2;
  __bf16* khb   = (__bf16*)p; p += (size_t)NH * S_TOT * HDP * 2;
  __bf16* vTb   = (__bf16*)p; p += (size_t)NH * HDP * S_TOT * 2;
  __bf16* attnb = (__bf16*)p; p += (size_t)S_TOT * EMB * 2;

  // 0) convert inputs to bf16
  cvt_bf16_kernel<<<2048, 256, 0, stream>>>(x, xbf, S_TOT * EMB);
  cvt_bf16_kernel<<<2048, 256, 0, stream>>>(w_qkv, wqkvb, 3 * EMB * EMB);
  cvt_bf16_kernel<<<2048, 256, 0, stream>>>(w_proj, wprjb, EMB * EMB);

  // 1) QKV projection: [3072, 3840] bf16
  {
    dim3 grid(3 * EMB / 128, S_TOT / 64);
    gemm_bf16_kernel<false><<<grid, 256, 0, stream>>>(
        xbf, wqkvb, b_qkv, (void*)qkvb, S_TOT, 3 * EMB, EMB);
  }

  // 2) RoPE + head-major / transposed-V layouts
  {
    const int total = S_TOT * NH * HDP;
    rope_kernel<<<(total + 255) / 256, 256, 0, stream>>>(qkvb, cosb, sinb,
                                                         qhb, khb, vTb);
  }

  // 3) block-diagonal attention
  attn_kernel<<<NH * NSEG * (SEG / 16), 256, 0, stream>>>(qhb, khb, vTb, attnb);

  // 4) output projection -> fp32 d_out
  {
    dim3 grid(EMB / 128, S_TOT / 64);
    gemm_bf16_kernel<true><<<grid, 256, 0, stream>>>(
        attnb, wprjb, b_proj, d_out, S_TOT, EMB, EMB);
  }
}
